// VectorQuantizer_2061584302597
// MI455X (gfx1250) — matrix operand order repaired and checked by reading, not compiled, not run on hardware
//
#include <hip/hip_runtime.h>

// ---------------------------------------------------------------------------
// VQ: quantized_st = codebook[argmin_k dist(latent, codebook_k)],
//     vq_loss = 1.25 * mean((q - l)^2)
// argmin via bf16 hi/lo split WMMA GEMM (score = dot - 0.5*||c||^2, argmax).
// Single accumulator chain: D->C WMMA chaining is full-rate (no NOPs needed),
// and keeping one acc avoids the register spills seen with 2-3 chains.
// ---------------------------------------------------------------------------

typedef __attribute__((ext_vector_type(16))) __bf16          v16bf;
typedef __attribute__((ext_vector_type(8)))  float           v8f;
typedef __attribute__((ext_vector_type(16))) unsigned short  v16us;
typedef __attribute__((ext_vector_type(8)))  unsigned short  v8us;

#define N_ROWS   32768
#define K_CB     4096
#define D_FEAT   256
#define TOTAL_EL (32768.0f * 256.0f)

// ---- workspace byte offsets ----
#define WS_CBHI   0u          // 4096*256 u16 = 2 MB  (bf16 hi plane)
#define WS_CBLO   2097152u    // 2 MB                 (bf16 lo plane)
#define WS_NORM   4194304u    // 4096 f32 = 16 KB     (-0.5*||c||^2)
#define WS_IDX    4210688u    // 32768 i32 = 128 KB
#define WS_PART   4341760u    // 4096 f32 = 16 KB     (per-block loss partials)

__device__ __forceinline__ unsigned short bf16_rne(float f) {
    unsigned int u = __float_as_uint(f);
    return (unsigned short)((u + 0x7FFFu + ((u >> 16) & 1u)) >> 16);
}
__device__ __forceinline__ void split_f32(float f, unsigned short& h, unsigned short& l) {
    unsigned short hb = bf16_rne(f);
    float hf = __uint_as_float(((unsigned int)hb) << 16);
    h = hb;
    l = bf16_rne(f - hf);
}

// ---------------------------------------------------------------------------
// Kernel 1: codebook -> bf16 hi/lo planes + exact -0.5*||c||^2 norms.
// One wave32 per codebook entry (8 floats per lane).
// ---------------------------------------------------------------------------
__global__ __launch_bounds__(256) void vq_cb_prep(const float* __restrict__ cb,
                                                  unsigned short* __restrict__ cbhi,
                                                  unsigned short* __restrict__ cblo,
                                                  float* __restrict__ cbnorm) {
    int tid  = threadIdx.x;
    int w    = tid >> 5;
    int lane = tid & 31;
    int e    = blockIdx.x * 8 + w;                 // 512 blocks * 8 = 4096 entries
    const float* rp = cb + (size_t)e * D_FEAT + lane * 8;
    float4 a = *(const float4*)(rp);
    float4 b = *(const float4*)(rp + 4);
    float f[8] = {a.x, a.y, a.z, a.w, b.x, b.y, b.z, b.w};
    v8us uh, ul;
    float ss = 0.f;
#pragma unroll
    for (int i = 0; i < 8; ++i) {
        unsigned short hh, ll;
        split_f32(f[i], hh, ll);
        uh[i] = hh; ul[i] = ll;
        ss += f[i] * f[i];
    }
    *(v8us*)(cbhi + (size_t)e * D_FEAT + lane * 8) = uh;
    *(v8us*)(cblo + (size_t)e * D_FEAT + lane * 8) = ul;
#pragma unroll
    for (int d = 1; d < 32; d <<= 1) ss += __shfl_xor(ss, d, 32);
    if (lane == 0) cbnorm[e] = -0.5f * ss;
}

// ---------------------------------------------------------------------------
// Kernel 2: fused GEMM + running argmax.
// Block = 256 threads = 8 waves, each wave owns a 16-row latent tile kept
// resident in registers (bf16 hi/lo A operands for all 8 k-steps).
// Codebook streamed in 16-entry tiles through double-buffered LDS laid out
// directly in WMMA B-operand order.
// ---------------------------------------------------------------------------
__global__ __launch_bounds__(256) void vq_argmin(const float* __restrict__ latents,
                                                 const unsigned short* __restrict__ cbhi,
                                                 const unsigned short* __restrict__ cblo,
                                                 const float* __restrict__ cbnorm,
                                                 int* __restrict__ idx_out) {
    // LDS: [buf(2)][plane(2)][j(8)][laneSlot(32)][elem(16)] bf16 = 32 KB
    __shared__ unsigned short smem[16384];

    const int tid  = threadIdx.x;
    const int w    = tid >> 5;
    const int lane = tid & 31;
    const int hx   = lane >> 4;          // lane half
    const int ln16 = lane & 15;
    const int row_base = blockIdx.x * 128 + w * 16;
    const int row      = row_base + ln16;

    // ---- Load + split the wave's 16x256 latent tile into A operands ----
    // A 16x32 bf16 layout: elem e -> K = (e%8) + (e/8)*16 + hx*8
    v16bf ahi[8], alo[8];
#pragma unroll
    for (int j = 0; j < 8; ++j) {
        const float* rp = latents + (size_t)row * D_FEAT + j * 32 + hx * 8;
        float4 s0 = *(const float4*)(rp);
        float4 s1 = *(const float4*)(rp + 4);
        float4 s2 = *(const float4*)(rp + 16);
        float4 s3 = *(const float4*)(rp + 20);
        float f[16] = {s0.x, s0.y, s0.z, s0.w, s1.x, s1.y, s1.z, s1.w,
                       s2.x, s2.y, s2.z, s2.w, s3.x, s3.y, s3.z, s3.w};
        v16us uh, ul;
#pragma unroll
        for (int e = 0; e < 16; ++e) {
            unsigned short hh, ll;
            split_f32(f[e], hh, ll);
            uh[e] = hh; ul[e] = ll;
        }
        ahi[j] = __builtin_bit_cast(v16bf, uh);
        alo[j] = __builtin_bit_cast(v16bf, ul);
    }

    float best[8];
    int   bidx[8];
#pragma unroll
    for (int r = 0; r < 8; ++r) { best[r] = -3.4e38f; bidx[r] = 0x7fffffff; }

    uint4 pre[4];

    // stage tile t's 16 entries (hi+lo planes, 16 KB) via 4 chunks/thread
    auto stage_load = [&](int t) {
        int cb0 = t * 16;
#pragma unroll
        for (int i = 0; i < 4; ++i) {
            int c     = tid + 256 * i;           // 0..1023
            int plane = c >> 9;
            int rem   = c & 511;
            int entry = rem >> 5;
            int f     = (rem & 31) * 8;          // feature start, mult of 8
            const unsigned short* src =
                (plane ? cblo : cbhi) + (size_t)(cb0 + entry) * D_FEAT + f;
            pre[i] = *(const uint4*)src;
        }
    };
    auto stage_store = [&](int t) {
        int bufo = (t & 1) * 8192;
#pragma unroll
        for (int i = 0; i < 4; ++i) {
            int c     = tid + 256 * i;
            int plane = c >> 9;
            int rem   = c & 511;
            int entry = rem >> 5;
            int f     = (rem & 31) * 8;
            int j     = f >> 5;
            int slot  = entry + ((f >> 3) & 1) * 16;
            int off   = bufo + plane * 4096 + j * 512 + slot * 16 + ((f >> 1) & 8);
            *(uint4*)&smem[off] = pre[i];
        }
    };

    stage_load(0);
    stage_store(0);
    __syncthreads();

    for (int t = 0; t < 256; ++t) {
        if (t + 1 < 256) stage_load(t + 1);

        int   cb_base = t * 16;
        float cn  = cbnorm[cb_base + ln16];
        int   col = cb_base + ln16;
        int   bufo = (t & 1) * 8192;

        v8f acc = {0.f, 0.f, 0.f, 0.f, 0.f, 0.f, 0.f, 0.f};
#pragma unroll
        for (int j = 0; j < 8; ++j) {
            const v16bf bhi = *(const v16bf*)&smem[bufo + j * 512 + lane * 16];
            const v16bf blo = *(const v16bf*)&smem[bufo + 4096 + j * 512 + lane * 16];
            acc = __builtin_amdgcn_wmma_f32_16x16x32_bf16(false, ahi[j], false, bhi,
                                                          (short)0, acc, false, false);
            acc = __builtin_amdgcn_wmma_f32_16x16x32_bf16(false, ahi[j], false, blo,
                                                          (short)0, acc, false, false);
            acc = __builtin_amdgcn_wmma_f32_16x16x32_bf16(false, alo[j], false, bhi,
                                                          (short)0, acc, false, false);
        }
#pragma unroll
        for (int r = 0; r < 8; ++r) {
            float s = acc[r] + cn;
            if (s > best[r]) { best[r] = s; bidx[r] = col; }
        }

        if (t + 1 < 256) stage_store(t + 1);
        __syncthreads();
    }

    // Cross-lane argmax within each 16-lane half (row m = r + 8*hx), with
    // smaller-index tie-break to mirror argmin-first semantics.
#pragma unroll
    for (int r = 0; r < 8; ++r) {
        float s = best[r];
        int   bi = bidx[r];
#pragma unroll
        for (int d = 1; d < 16; d <<= 1) {
            float os = __shfl_xor(s, d, 32);
            int   oi = __shfl_xor(bi, d, 32);
            if (os > s || (os == s && oi < bi)) { s = os; bi = oi; }
        }
        if (ln16 == 0) idx_out[row_base + hx * 8 + r] = bi;
    }
}

// ---------------------------------------------------------------------------
// Kernel 3: gather codebook[idx] -> out, per-block SSE partials (deterministic).
// One wave per row, 8 rows per block -> 4096 blocks.
// ---------------------------------------------------------------------------
__global__ __launch_bounds__(256) void vq_gather(const float* __restrict__ latents,
                                                 const float* __restrict__ cb,
                                                 const int* __restrict__ idx,
                                                 float* __restrict__ out,
                                                 float* __restrict__ part) {
    __shared__ float wsum[8];
    int tid  = threadIdx.x;
    int w    = tid >> 5;
    int lane = tid & 31;
    int row  = blockIdx.x * 8 + w;
    int id   = idx[row];

    const float* qp = cb + (size_t)id * D_FEAT + lane * 8;
    const float* lp = latents + (size_t)row * D_FEAT + lane * 8;
    float4 q0 = *(const float4*)(qp);
    float4 q1 = *(const float4*)(qp + 4);
    float4 l0 = *(const float4*)(lp);
    float4 l1 = *(const float4*)(lp + 4);
    *(float4*)(out + (size_t)row * D_FEAT + lane * 8)     = q0;
    *(float4*)(out + (size_t)row * D_FEAT + lane * 8 + 4) = q1;

    float d0 = q0.x - l0.x, d1 = q0.y - l0.y, d2 = q0.z - l0.z, d3 = q0.w - l0.w;
    float d4 = q1.x - l1.x, d5 = q1.y - l1.y, d6 = q1.z - l1.z, d7 = q1.w - l1.w;
    float s = d0*d0 + d1*d1 + d2*d2 + d3*d3 + d4*d4 + d5*d5 + d6*d6 + d7*d7;
#pragma unroll
    for (int d = 1; d < 32; d <<= 1) s += __shfl_xor(s, d, 32);
    if (lane == 0) wsum[w] = s;
    __syncthreads();
    if (tid == 0) {
        float b = 0.f;
#pragma unroll
        for (int i = 0; i < 8; ++i) b += wsum[i];   // fixed order
        part[blockIdx.x] = b;
    }
}

// ---------------------------------------------------------------------------
// Kernel 4: deterministic final reduction of 4096 partials -> vq_loss.
// ---------------------------------------------------------------------------
__global__ __launch_bounds__(256) void vq_final(const float* __restrict__ part,
                                                float* __restrict__ out) {
    __shared__ float red[256];
    int tid = threadIdx.x;
    float s = 0.f;
#pragma unroll
    for (int i = 0; i < 16; ++i) s += part[i * 256 + tid];  // fixed order
    red[tid] = s;
    __syncthreads();
    for (int off = 128; off > 0; off >>= 1) {
        if (tid < off) red[tid] += red[tid + off];
        __syncthreads();
    }
    if (tid == 0) out[(size_t)N_ROWS * D_FEAT] = 1.25f * red[0] / TOTAL_EL;
}

// ---------------------------------------------------------------------------
extern "C" void kernel_launch(void* const* d_in, const int* in_sizes, int n_in,
                              void* d_out, int out_size, void* d_ws, size_t ws_size,
                              hipStream_t stream) {
    (void)in_sizes; (void)n_in; (void)out_size; (void)ws_size;
    const float* latents  = (const float*)d_in[0];
    const float* codebook = (const float*)d_in[1];
    float* out = (float*)d_out;
    char*  ws  = (char*)d_ws;

    unsigned short* cbhi   = (unsigned short*)(ws + WS_CBHI);
    unsigned short* cblo   = (unsigned short*)(ws + WS_CBLO);
    float*          cbnorm = (float*)(ws + WS_NORM);
    int*            idx    = (int*)(ws + WS_IDX);
    float*          part   = (float*)(ws + WS_PART);

    vq_cb_prep<<<512, 256, 0, stream>>>(codebook, cbhi, cblo, cbnorm);
    vq_argmin<<<256, 256, 0, stream>>>(latents, cbhi, cblo, cbnorm, idx);
    vq_gather<<<4096, 256, 0, stream>>>(latents, codebook, idx, out, part);
    vq_final<<<1, 256, 0, stream>>>(part, out);
}
